// SchNetFlowMatching_54039278518701
// MI455X (gfx1250) — matrix operand order repaired and checked by reading, compiled, not run on hardware
//
#include <hip/hip_runtime.h>

// ---------------------------------------------------------------------------
// SchNet flow-matching forward, CDNA5 (gfx1250), wave32 + WMMA f16.
//   - weights packed once: fp32 -> transposed/K-padded f16 (WMMA-B layout)
//   - edge kernel (hot, E=1.6M): gaussians in regs -> WMMA(64->64->64) with
//     register-resident weight fragments -> *x[src] -> atomicAdd agg[dst]
//   - node kernels: A-fragments straight from global f32; B-fragments batch-
//     preloaded per n-tile (1 wait -> 4 WMMAs); nt=8 loops kept rolled
//     (#pragma unroll 1) to bound VGPR pressure -- no scratch spills
//   - tile-level full/partial guards (scalar branch, no per-element EXEC churn)
// ---------------------------------------------------------------------------

typedef __attribute__((ext_vector_type(16))) _Float16 v16h;
typedef __attribute__((ext_vector_type(8)))  float    v8f;

#define NGAU 50
#define NLAY 4

static __device__ __forceinline__ float fsilu(float v) {
  return v * (1.0f / (1.0f + __expf(-v)));
}

static __device__ __forceinline__ v8f wmma16(v16h a, v16h b, v8f c) {
  return __builtin_amdgcn_wmma_f32_16x16x32_f16(false, a, false, b, (short)0, c,
                                                false, false);
}

// A-fragment from row-major f16 [16][ldk] (LDS staging buffers).
static __device__ __forceinline__ v16h fragA(const _Float16* base, int ldk,
                                             int kbase) {
  int lane = threadIdx.x & 31;
  int row  = lane & 15;
  int hi8  = (lane >> 4) << 3;
  v16h a;
#pragma unroll
  for (int p = 0; p < 8; ++p) {
    int k = kbase + 2 * p + ((p >= 4) ? 8 : 0) + hi8;
    union { unsigned int u; _Float16 h[2]; } cv;
    cv.u = *(const unsigned int*)(base + row * ldk + k);
    a[2 * p]     = cv.h[0];
    a[2 * p + 1] = cv.h[1];
  }
  return a;
}

// A-fragment directly from a per-lane global f32 row pointer (converts on load).
static __device__ __forceinline__ v16h fragA_f32(const float* rowptr, int kbase) {
  int lane = threadIdx.x & 31;
  int hi8  = (lane >> 4) << 3;
  v16h a;
#pragma unroll
  for (int p = 0; p < 8; ++p) {
    int k = kbase + 2 * p + ((p >= 4) ? 8 : 0) + hi8;
    float2 f = *(const float2*)(rowptr + k);
    a[2 * p]     = (_Float16)f.x;
    a[2 * p + 1] = (_Float16)f.y;
  }
  return a;
}

// B-fragment from transposed f16 weights Wt[dout][ldk] (global, L2-resident).
static __device__ __forceinline__ v16h fragB(const _Float16* wt, int ldk,
                                             int ntile, int kbase) {
  int lane = threadIdx.x & 31;
  int n    = (ntile << 4) + (lane & 15);
  int hi8  = (lane >> 4) << 3;
  v16h b;
#pragma unroll
  for (int p = 0; p < 8; ++p) {
    int k = kbase + 2 * p + ((p >= 4) ? 8 : 0) + hi8;
    union { unsigned int u; _Float16 h[2]; } cv;
    cv.u = *(const unsigned int*)(wt + n * ldk + k);
    b[2 * p]     = cv.h[0];
    b[2 * p + 1] = cv.h[1];
  }
  return b;
}

// ---- weight pack: W(din x dout, row-major f32) -> Wt[dout][dinpad] f16 ----
__global__ void pack_wt(const float* __restrict__ W, _Float16* __restrict__ Wt,
                        int din, int dout, int dinpad) {
  int i = blockIdx.x * blockDim.x + threadIdx.x;
  if (i >= dout * dinpad) return;
  int n = i / dinpad, k = i % dinpad;
  Wt[i] = (_Float16)((k < din) ? W[k * dout + n] : 0.0f);
}

// ---- time + condition embeddings (B=16, scalar) ----
__global__ void embed_bc_kernel(const float* t, const float* y,
                                const float* tW0, const float* tb0,
                                const float* tW1, const float* tb1,
                                const float* cW0, const float* cb0,
                                const float* cW1, const float* cb1,
                                float* comb, int B) {
  __shared__ float emb[64];
  __shared__ float hid[128];
  __shared__ float acc[128];
  int tid = threadIdx.x;  // 128 threads
  for (int b = 0; b < B; ++b) {
    for (int pass = 0; pass < 2; ++pass) {
      float val = (pass == 0) ? t[b] : (y[b] - 1.0f) * (1.0f / 3.0f);
      if (tid < 32) {
        float fr = __expf(-logf(10000.0f) * (float)tid * (1.0f / 32.0f));
        float a  = val * fr;
        emb[tid]      = __sinf(a);
        emb[tid + 32] = __cosf(a);
      }
      __syncthreads();
      const float* W0 = pass ? cW0 : tW0;
      const float* b0 = pass ? cb0 : tb0;
      const float* W1 = pass ? cW1 : tW1;
      const float* b1 = pass ? cb1 : tb1;
      float s = b0[tid];
      for (int k = 0; k < 64; ++k) s += emb[k] * W0[k * 128 + tid];
      hid[tid] = fsilu(s);
      __syncthreads();
      float o = b1[tid];
      for (int i = 0; i < 128; ++i) o += hid[i] * W1[i * 128 + tid];
      if (pass == 0) acc[tid] = o;
      else           comb[b * 128 + tid] = acc[tid] + o;
      __syncthreads();
    }
  }
}

// ---- node embedding: h = MLP2(pos) + comb[batch] ----
__global__ void node_embed_kernel(const float* __restrict__ pos,
                                  const int* __restrict__ batch,
                                  const float* __restrict__ W0,   // 3x128 f32
                                  const float* __restrict__ b0,
                                  const _Float16* __restrict__ Wt1,  // [128][128]
                                  const float* __restrict__ b1,
                                  const float* __restrict__ comb,
                                  float* __restrict__ h, int N) {
  __shared__ _Float16 hidBuf[4][16 * 128];
  int wslot = threadIdx.x >> 5, lane = threadIdx.x & 31;
  int wave = blockIdx.x * (blockDim.x >> 5) + wslot;
  int nw   = gridDim.x * (blockDim.x >> 5);
  int nTiles = (N + 15) >> 4;
  _Float16* hb = hidBuf[wslot];
  int mhi = (lane >> 4) << 3, colb = lane & 15;

  for (int tile = wave; tile < nTiles; tile += nw) {
    int n0 = tile << 4;
    for (int i = lane; i < 16 * 128; i += 32) {
      int row = i >> 7, c = i & 127;
      int node = min(n0 + row, N - 1);
      float s = b0[c] + pos[node * 3 + 0] * W0[c] +
                pos[node * 3 + 1] * W0[128 + c] +
                pos[node * 3 + 2] * W0[256 + c];
      hb[i] = (_Float16)fsilu(s);
    }
    v16h aF[4];
#pragma unroll
    for (int ks = 0; ks < 4; ++ks) aF[ks] = fragA(hb, 128, ks * 32);

#pragma unroll 1
    for (int nt = 0; nt < 8; ++nt) {
      v16h bf[4];
#pragma unroll
      for (int ks = 0; ks < 4; ++ks) bf[ks] = fragB(Wt1, 128, nt, ks * 32);
      v8f acc = {};
#pragma unroll
      for (int ks = 0; ks < 4; ++ks) acc = wmma16(aF[ks], bf[ks], acc);
      int col = nt * 16 + colb;
      float bias = b1[col];
      if (n0 + 16 <= N) {
#pragma unroll
        for (int r = 0; r < 8; ++r) {
          int node = n0 + r + mhi;
          h[node * 128 + col] = acc[r] + bias + comb[batch[node] * 128 + col];
        }
      } else {
        for (int r = 0; r < 8; ++r) {
          int node = n0 + r + mhi;
          if (node < N)
            h[node * 128 + col] = acc[r] + bias + comb[batch[node] * 128 + col];
        }
      }
    }
  }
}

// ---- edge distances (once) ----
__global__ void dist_kernel(const float* __restrict__ pos,
                            const int* __restrict__ ei,
                            float* __restrict__ dist, int E) {
  int i = blockIdx.x * blockDim.x + threadIdx.x;
  if (i >= E) return;
  int s = ei[i], d = ei[E + i];
  float dx = pos[d * 3 + 0] - pos[s * 3 + 0];
  float dy = pos[d * 3 + 1] - pos[s * 3 + 1];
  float dz = pos[d * 3 + 2] - pos[s * 3 + 2];
  dist[i] = sqrtf(dx * dx + dy * dy + dz * dz);
}

__global__ void zero_kernel(float* p, int n) {
  int i = blockIdx.x * blockDim.x + threadIdx.x;
  if (i < n) p[i] = 0.0f;
}

// ---- x = MLP2(h): 128 -> 64 -> 64 ----
__global__ void atom_mlp_kernel(const float* __restrict__ h,
                                const _Float16* __restrict__ Wt0,  // [64][128]
                                const float* __restrict__ b0,
                                const _Float16* __restrict__ Wt1,  // [64][64]
                                const float* __restrict__ b1,
                                float* __restrict__ x, int N) {
  __shared__ _Float16 hidBuf[4][16 * 64];
  int wslot = threadIdx.x >> 5, lane = threadIdx.x & 31;
  int wave = blockIdx.x * (blockDim.x >> 5) + wslot;
  int nw   = gridDim.x * (blockDim.x >> 5);
  int nTiles = (N + 15) >> 4;
  _Float16* hb = hidBuf[wslot];
  int mhi = (lane >> 4) << 3, colb = lane & 15, row = lane & 15;

  // GEMM2 weight fragments are wave-invariant: keep in registers.
  v16h wb1[4][2];
  float rb0[4], rb1[4];
#pragma unroll
  for (int nt = 0; nt < 4; ++nt) {
    wb1[nt][0] = fragB(Wt1, 64, nt, 0);
    wb1[nt][1] = fragB(Wt1, 64, nt, 32);
    rb0[nt] = b0[nt * 16 + colb];
    rb1[nt] = b1[nt * 16 + colb];
  }

  for (int tile = wave; tile < nTiles; tile += nw) {
    int n0 = tile << 4;
    const float* rowp = h + min(n0 + row, N - 1) * 128;
    v16h aF[4];
#pragma unroll
    for (int ks = 0; ks < 4; ++ks) aF[ks] = fragA_f32(rowp, ks * 32);

#pragma unroll
    for (int nt = 0; nt < 4; ++nt) {
      v16h bf[4];
#pragma unroll
      for (int ks = 0; ks < 4; ++ks) bf[ks] = fragB(Wt0, 128, nt, ks * 32);
      v8f acc = {};
#pragma unroll
      for (int ks = 0; ks < 4; ++ks) acc = wmma16(aF[ks], bf[ks], acc);
      int col = nt * 16 + colb;
#pragma unroll
      for (int r = 0; r < 8; ++r)
        hb[(r + mhi) * 64 + col] = (_Float16)fsilu(acc[r] + rb0[nt]);
    }
    v16h hF0 = fragA(hb, 64, 0);
    v16h hF1 = fragA(hb, 64, 32);
#pragma unroll
    for (int nt = 0; nt < 4; ++nt) {
      v8f acc = {};
      acc = wmma16(hF0, wb1[nt][0], acc);
      acc = wmma16(hF1, wb1[nt][1], acc);
      int col = nt * 16 + colb;
      if (n0 + 16 <= N) {
#pragma unroll
        for (int r = 0; r < 8; ++r)
          x[(n0 + r + mhi) * 64 + col] = acc[r] + rb1[nt];
      } else {
        for (int r = 0; r < 8; ++r)
          if (n0 + r + mhi < N) x[(n0 + r + mhi) * 64 + col] = acc[r] + rb1[nt];
      }
    }
  }
}

// ---- fused edge kernel: gaussians -> MLP2(filt) -> * x[src] -> atomic agg[dst]
__global__ void edge_kernel(const int* __restrict__ ei,
                            const float* __restrict__ dist,
                            const _Float16* __restrict__ Wt0,  // [64][64] K pad 50->64
                            const float* __restrict__ b0,
                            const _Float16* __restrict__ Wt1,  // [64][64]
                            const float* __restrict__ b1,
                            const float* __restrict__ x,
                            float* __restrict__ agg, int E) {
  __shared__ _Float16 hidBuf[4][16 * 64];
  int wslot = threadIdx.x >> 5, lane = threadIdx.x & 31;
  int wave = blockIdx.x * (blockDim.x >> 5) + wslot;
  int nw   = gridDim.x * (blockDim.x >> 5);
  int nTiles = (E + 15) >> 4;
  _Float16* hb = hidBuf[wslot];
  int mhi = (lane >> 4) << 3, colb = lane & 15;

  // All weight fragments + biases are wave-invariant: registers only.
  v16h wb0[4][2], wb1[4][2];
  float rb0[4], rb1[4];
#pragma unroll
  for (int nt = 0; nt < 4; ++nt) {
    wb0[nt][0] = fragB(Wt0, 64, nt, 0);
    wb0[nt][1] = fragB(Wt0, 64, nt, 32);
    wb1[nt][0] = fragB(Wt1, 64, nt, 0);
    wb1[nt][1] = fragB(Wt1, 64, nt, 32);
    rb0[nt] = b0[nt * 16 + colb];
    rb1[nt] = b1[nt * 16 + colb];
  }

  const float delta = 5.0f / 49.0f;              // CUTOFF/(NG-1)
  const float coeff = -0.5f / (delta * delta);

  for (int tile = wave; tile < nTiles; tile += nw) {
    int e0 = tile << 4;
    float dv = 0.0f; int sv = 0, tv = 0;
    if (lane < 16) {
      int e = min(e0 + lane, E - 1);
      dv = dist[e];
      sv = ei[e];
      tv = ei[E + e];
    }
    if (tile + nw < nTiles)
      __builtin_prefetch(dist + ((tile + nw) << 4), 0, 0);

    float d = __shfl(dv, lane & 15, 32);
    // Gaussian smearing directly into WMMA A-fragments (K padded to 64)
    v16h aF[2];
#pragma unroll
    for (int ks = 0; ks < 2; ++ks) {
#pragma unroll
      for (int e = 0; e < 16; ++e) {
        int k = ks * 32 + e + ((e >= 8) ? 8 : 0) + mhi;
        float dd = d - (float)k * delta;
        float g = (k < NGAU) ? __expf(coeff * dd * dd) : 0.0f;
        aF[ks][e] = (_Float16)g;
      }
    }
    // GEMM1: (16x64)x(64x64), bias+SiLU -> per-wave LDS (relayout)
#pragma unroll
    for (int nt = 0; nt < 4; ++nt) {
      v8f acc = {};
      acc = wmma16(aF[0], wb0[nt][0], acc);
      acc = wmma16(aF[1], wb0[nt][1], acc);
      int col = nt * 16 + colb;
#pragma unroll
      for (int r = 0; r < 8; ++r)
        hb[(r + mhi) * 64 + col] = (_Float16)fsilu(acc[r] + rb0[nt]);
    }
    v16h hF0 = fragA(hb, 64, 0);
    v16h hF1 = fragA(hb, 64, 32);
    // GEMM2 entirely in registers
    v8f acc2[4];
#pragma unroll
    for (int nt = 0; nt < 4; ++nt) {
      v8f a = {};
      a = wmma16(hF0, wb1[nt][0], a);
      a = wmma16(hF1, wb1[nt][1], a);
      acc2[nt] = a;
    }
    int sArr[8], tArr[8];
#pragma unroll
    for (int r = 0; r < 8; ++r) {
      sArr[r] = __shfl(sv, r + mhi, 32) * 64;
      tArr[r] = __shfl(tv, r + mhi, 32) * 64;
    }
    // fused elementwise * x[src] + scatter-add (L2-resident), row-contiguous
    if (e0 + 16 <= E) {
#pragma unroll
      for (int r = 0; r < 8; ++r) {
#pragma unroll
        for (int nt = 0; nt < 4; ++nt) {
          int col = nt * 16 + colb;
          float w = acc2[nt][r] + rb1[nt];
          atomicAdd(&agg[tArr[r] + col], w * x[sArr[r] + col]);
        }
      }
    } else {
      for (int r = 0; r < 8; ++r) {
        if (e0 + r + mhi < E) {
          for (int nt = 0; nt < 4; ++nt) {
            int col = nt * 16 + colb;
            float w = acc2[nt][r] + rb1[nt];
            atomicAdd(&agg[tArr[r] + col], w * x[sArr[r] + col]);
          }
        }
      }
    }
  }
}

// ---- h += MLP2(agg): 64 -> 128 -> 128 ----
__global__ void out_mlp_kernel(const float* __restrict__ agg,
                               const _Float16* __restrict__ Wt0,  // [128][64]
                               const float* __restrict__ b0,
                               const _Float16* __restrict__ Wt1,  // [128][128]
                               const float* __restrict__ b1,
                               float* __restrict__ h, int N) {
  __shared__ _Float16 hidBuf[4][16 * 128];
  int wslot = threadIdx.x >> 5, lane = threadIdx.x & 31;
  int wave = blockIdx.x * (blockDim.x >> 5) + wslot;
  int nw   = gridDim.x * (blockDim.x >> 5);
  int nTiles = (N + 15) >> 4;
  _Float16* hb = hidBuf[wslot];
  int mhi = (lane >> 4) << 3, colb = lane & 15, row = lane & 15;

  for (int tile = wave; tile < nTiles; tile += nw) {
    int n0 = tile << 4;
    const float* rowp = agg + min(n0 + row, N - 1) * 64;
    v16h aF0 = fragA_f32(rowp, 0);
    v16h aF1 = fragA_f32(rowp, 32);
#pragma unroll 1
    for (int nt = 0; nt < 8; ++nt) {
      v16h bf0 = fragB(Wt0, 64, nt, 0);
      v16h bf1 = fragB(Wt0, 64, nt, 32);
      v8f acc = {};
      acc = wmma16(aF0, bf0, acc);
      acc = wmma16(aF1, bf1, acc);
      int col = nt * 16 + colb;
      float bias = b0[col];
#pragma unroll
      for (int r = 0; r < 8; ++r)
        hb[(r + mhi) * 128 + col] = (_Float16)fsilu(acc[r] + bias);
    }
    v16h hF[4];
#pragma unroll
    for (int ks = 0; ks < 4; ++ks) hF[ks] = fragA(hb, 128, ks * 32);
#pragma unroll 1
    for (int nt = 0; nt < 8; ++nt) {
      v16h bf[4];
#pragma unroll
      for (int ks = 0; ks < 4; ++ks) bf[ks] = fragB(Wt1, 128, nt, ks * 32);
      v8f acc = {};
#pragma unroll
      for (int ks = 0; ks < 4; ++ks) acc = wmma16(hF[ks], bf[ks], acc);
      int col = nt * 16 + colb;
      float bias = b1[col];
      if (n0 + 16 <= N) {
#pragma unroll
        for (int r = 0; r < 8; ++r)
          h[(n0 + r + mhi) * 128 + col] += acc[r] + bias;
      } else {
        for (int r = 0; r < 8; ++r)
          if (n0 + r + mhi < N) h[(n0 + r + mhi) * 128 + col] += acc[r] + bias;
      }
    }
  }
}

// ---- velocity = MLP2(h): 128 -> 128 -> 3 ----
__global__ void output_kernel(const float* __restrict__ h,
                              const _Float16* __restrict__ Wt0,  // [128][128]
                              const float* __restrict__ b0,
                              const float* __restrict__ W1,      // 128x3 f32
                              const float* __restrict__ b1,
                              float* __restrict__ out, int N) {
  __shared__ float hidF[4][16 * 128];
  int wslot = threadIdx.x >> 5, lane = threadIdx.x & 31;
  int wave = blockIdx.x * (blockDim.x >> 5) + wslot;
  int nw   = gridDim.x * (blockDim.x >> 5);
  int nTiles = (N + 15) >> 4;
  float* hf = hidF[wslot];
  int mhi = (lane >> 4) << 3, colb = lane & 15, row = lane & 15;

  for (int tile = wave; tile < nTiles; tile += nw) {
    int n0 = tile << 4;
    const float* rowp = h + min(n0 + row, N - 1) * 128;
    v16h aF[4];
#pragma unroll
    for (int ks = 0; ks < 4; ++ks) aF[ks] = fragA_f32(rowp, ks * 32);
#pragma unroll 1
    for (int nt = 0; nt < 8; ++nt) {
      v16h bf[4];
#pragma unroll
      for (int ks = 0; ks < 4; ++ks) bf[ks] = fragB(Wt0, 128, nt, ks * 32);
      v8f acc = {};
#pragma unroll
      for (int ks = 0; ks < 4; ++ks) acc = wmma16(aF[ks], bf[ks], acc);
      int col = nt * 16 + colb;
      float bias = b0[col];
#pragma unroll
      for (int r = 0; r < 8; ++r)
        hf[(r + mhi) * 128 + col] = fsilu(acc[r] + bias);
    }
    // final 128 -> 3 (48 outputs per tile)
    for (int i = lane; i < 48; i += 32) {
      int rr = i / 3, j = i % 3;
      int node = n0 + rr;
      if (node < N) {
        float s = b1[j];
        for (int k = 0; k < 128; ++k) s += hf[rr * 128 + k] * W1[k * 3 + j];
        out[node * 3 + j] = s;
      }
    }
  }
}

// ---------------------------------------------------------------------------
extern "C" void kernel_launch(void* const* d_in, const int* in_sizes, int n_in,
                              void* d_out, int out_size, void* d_ws,
                              size_t ws_size, hipStream_t stream) {
  (void)n_in; (void)out_size; (void)ws_size;
  const int N = in_sizes[0] / 3;
  const int E = in_sizes[1] / 2;
  const int B = in_sizes[3];

  auto F = [&](int i) { return (const float*)d_in[i]; };
  const float* pos   = F(0);
  const int*   ei    = (const int*)d_in[1];
  const int*   batch = (const int*)d_in[2];
  const float* t     = F(3);
  const float* y     = F(4);
  // params in jax-pytree (sorted-key) order:
  // atom_embed, cond_mlp, layers[0..3]{atom,filt,out}, output, time_mlp
  const float *aeW0 = F(5), *aeB0 = F(6), *aeW1 = F(7), *aeB1 = F(8);
  const float *cW0 = F(9), *cB0 = F(10), *cW1 = F(11), *cB1 = F(12);
  const float *oW0 = F(61), *oB0 = F(62), *oW1 = F(63), *oB1 = F(64);
  const float *tW0 = F(65), *tB0 = F(66), *tW1 = F(67), *tB1 = F(68);

  char* ws = (char*)d_ws;
  auto alloc = [&](size_t bytes) {
    void* p = ws;
    ws += (bytes + 255) & ~(size_t)255;
    return p;
  };
  float* comb = (float*)alloc((size_t)B * 128 * 4);
  float* h    = (float*)alloc((size_t)N * 128 * 4);
  float* x    = (float*)alloc((size_t)N * 64 * 4);
  float* agg  = (float*)alloc((size_t)N * 64 * 4);
  float* dist = (float*)alloc((size_t)E * 4);

  _Float16* pAE1  = (_Float16*)alloc(128 * 128 * 2);
  _Float16* pOut0 = (_Float16*)alloc(128 * 128 * 2);
  _Float16 *pA0[NLAY], *pA1[NLAY], *pF0[NLAY], *pF1[NLAY], *pO0[NLAY], *pO1[NLAY];
  for (int l = 0; l < NLAY; ++l) {
    pA0[l] = (_Float16*)alloc(64 * 128 * 2);
    pA1[l] = (_Float16*)alloc(64 * 64 * 2);
    pF0[l] = (_Float16*)alloc(64 * 64 * 2);
    pF1[l] = (_Float16*)alloc(64 * 64 * 2);
    pO0[l] = (_Float16*)alloc(128 * 64 * 2);
    pO1[l] = (_Float16*)alloc(128 * 128 * 2);
  }

  auto pack = [&](const float* W, _Float16* dst, int din, int dout, int pad) {
    int total = dout * pad;
    pack_wt<<<(total + 255) / 256, 256, 0, stream>>>(W, dst, din, dout, pad);
  };
  pack(aeW1, pAE1, 128, 128, 128);
  pack(oW0, pOut0, 128, 128, 128);
  for (int l = 0; l < NLAY; ++l) {
    int Lb = 13 + 12 * l;
    pack(F(Lb + 0),  pA0[l], 128, 64, 128);  // atom.0
    pack(F(Lb + 2),  pA1[l], 64, 64, 64);    // atom.1
    pack(F(Lb + 4),  pF0[l], 50, 64, 64);    // filt.0 (K padded 50->64)
    pack(F(Lb + 6),  pF1[l], 64, 64, 64);    // filt.1
    pack(F(Lb + 8),  pO0[l], 64, 128, 64);   // out.0
    pack(F(Lb + 10), pO1[l], 128, 128, 128); // out.1
  }

  embed_bc_kernel<<<1, 128, 0, stream>>>(t, y, tW0, tB0, tW1, tB1, cW0, cB0,
                                         cW1, cB1, comb, B);
  node_embed_kernel<<<256, 128, 0, stream>>>(pos, batch, aeW0, aeB0, pAE1,
                                             aeB1, comb, h, N);
  dist_kernel<<<(E + 255) / 256, 256, 0, stream>>>(pos, ei, dist, E);

  for (int l = 0; l < NLAY; ++l) {
    int Lb = 13 + 12 * l;
    zero_kernel<<<(N * 64 + 255) / 256, 256, 0, stream>>>(agg, N * 64);
    atom_mlp_kernel<<<256, 128, 0, stream>>>(h, pA0[l], F(Lb + 1), pA1[l],
                                             F(Lb + 3), x, N);
    edge_kernel<<<1024, 128, 0, stream>>>(ei, dist, pF0[l], F(Lb + 5), pF1[l],
                                          F(Lb + 7), x, agg, E);
    out_mlp_kernel<<<256, 128, 0, stream>>>(agg, pO0[l], F(Lb + 9), pO1[l],
                                            F(Lb + 11), h, N);
  }

  output_kernel<<<256, 128, 0, stream>>>(h, pOut0, oB0, oW1, oB1,
                                         (float*)d_out, N);
}
